// RNNCell_6854767805226
// MI455X (gfx1250) — matrix operand order repaired and checked by reading, not compiled, not run on hardware
//
#include <hip/hip_runtime.h>

// CDNA5 / gfx1250 wave32 WMMA types
typedef __attribute__((ext_vector_type(16))) __bf16        v16bf;
typedef __attribute__((ext_vector_type(8)))  float         v8f;
typedef __attribute__((ext_vector_type(4)))  unsigned int  v4u;

union Frag16 {
    v16bf v;
    v4u   q[2];
    unsigned int u[8];
};

#define MDIM 64

// ---------------------------------------------------------------------------
// Setup: odot = (P - I) * mu, stored transposed (odotT[n][k]) as bf16 so the
// main kernel can load WMMA B-fragments with contiguous 128-bit reads.
// P[k][n] (n != k) = data_p[k][n - (n>k)] / rowsum(data_p[k]).
// ---------------------------------------------------------------------------
__global__ void build_odot_kernel(const float* __restrict__ mu,
                                  const float* __restrict__ data_p,
                                  unsigned short* __restrict__ odotT) {
    int k = blockIdx.x * blockDim.x + threadIdx.x;
    if (k >= MDIM) return;
    float s = 0.0f;
    for (int j = 0; j < MDIM - 1; ++j) s += data_p[k * (MDIM - 1) + j];
    float inv = 1.0f / s;
    float muk = mu[k];
    for (int n = 0; n < MDIM; ++n) {
        float p;
        if (n == k) p = -1.0f;
        else        p = data_p[k * (MDIM - 1) + (n - (n > k ? 1 : 0))] * inv;
        float val = p * muk;
        odotT[n * MDIM + k] = __builtin_bit_cast(unsigned short, (__bf16)val);
    }
}

// ---------------------------------------------------------------------------
// Main scan. One wave32 per 16 batch rows. odot resident in VGPRs as 8 bf16
// B-fragments (2 K-blocks x 4 N-tiles). jobs state in 4 f32 C/D fragments.
// Per step: 8x v_wmma_f32_16x16x32_bf16 (K=64 contraction, N=64 output),
// A staged via LDS and gathered with 4x ds_load_b128, streaming I/O with
// non-temporal hints.
// ---------------------------------------------------------------------------
__global__ void __launch_bounds__(32, 1)
rnn_scan_kernel(const float* __restrict__ in,   // (B,T,65)
                const float* __restrict__ h0,   // (B,65)
                const float* __restrict__ trace,// (64)
                const unsigned short* __restrict__ odotT, // (64,64) bf16, [n][k]
                float* __restrict__ out,        // (B,T,65)
                int T) {
    __shared__ alignas(16) unsigned short Amat[16 * MDIM]; // bf16 A staging
    __shared__ float st[16];  // state time per row
    __shared__ float ct[16];  // cur_t per row
    __shared__ float fl[16];  // x[:,1] per row

    const int lane = threadIdx.x;         // 0..31
    const int half = lane >> 4;           // 0 or 1
    const int lcol = lane & 15;           // 0..15
    const int b0   = blockIdx.x * 16;

    // ---- Load resident B fragments (dense bf16 32x16 layout):
    // lane half h: VGPRs 0-3 hold K=8h..8h+7, VGPRs 4-7 hold K=16+8h..16+8h+7 (N=lcol).
    // Per fragment: two runs of 8 contiguous bf16 per lane (K 16 apart) -> 2x 128-bit loads.
    Frag16 Bf[2][4];
    {
        const v4u* od128 = (const v4u*)odotT;
        #pragma unroll
        for (int n = 0; n < 4; ++n) {
            int ncol = n * 16 + lcol;
            #pragma unroll
            for (int kb = 0; kb < 2; ++kb) {
                int base = ncol * 8 + kb * 4 + half; // 16-byte units
                Bf[kb][n].q[0] = od128[base + 0];
                Bf[kb][n].q[1] = od128[base + 2];
            }
        }
    }

    float traceReg[4];
    #pragma unroll
    for (int n = 0; n < 4; ++n) traceReg[n] = trace[n * 16 + lcol];

    // ---- Init jobs state (C/D layout: lane, VGPR r -> row m = 8*half + r, col = n*16+lcol)
    v8f c[4];
    size_t rowOff[8];
    #pragma unroll
    for (int r = 0; r < 8; ++r) {
        int m = half * 8 + r;
        rowOff[r] = (size_t)(b0 + m) * (size_t)T * 65u;
        #pragma unroll
        for (int n = 0; n < 4; ++n)
            c[n][r] = h0[(size_t)(b0 + m) * 65u + 1 + n * 16 + lcol];
    }
    const size_t myRow = (size_t)(b0 + lane) * (size_t)T * 65u; // valid for lane<16
    if (lane < 16) st[lane] = h0[(size_t)(b0 + lane) * 65u];
    __syncthreads();

    const v4u* A128 = (const v4u*)Amat;

    for (int t = 0; t < T; ++t) {
        const size_t tOff = (size_t)t * 65u;

        // per-row scalars for this step (same cachelines as the xj loads below)
        if (lane < 16) {
            ct[lane] = __builtin_nontemporal_load(in + myRow + tOff + 0);
            fl[lane] = __builtin_nontemporal_load(in + myRow + tOff + 1);
            if (t + 1 < T) {
                __builtin_prefetch(in + myRow + tOff + 65, 0, 1);
                __builtin_prefetch(in + myRow + tOff + 65 + 32, 0, 1);
            }
        }
        __syncthreads();

        float dif[8], flg[8];
        #pragma unroll
        for (int r = 0; r < 8; ++r) {
            int m = half * 8 + r;
            dif[r] = ct[m] - st[m];
            flg[r] = fl[m];
        }

        // stage A = bf16(min(jobs, trace)) into LDS (row-major 16x64)
        #pragma unroll
        for (int n = 0; n < 4; ++n) {
            int coln = n * 16 + lcol;
            #pragma unroll
            for (int r = 0; r < 8; ++r) {
                int m = half * 8 + r;
                float a = c[n][r] < traceReg[n] ? c[n][r] : traceReg[n];
                Amat[m * MDIM + coln] = __builtin_bit_cast(unsigned short, (__bf16)a);
            }
        }
        __syncthreads();

        if (lane < 16) {
            st[lane] = ct[lane];
            __builtin_nontemporal_store(ct[lane], out + myRow + tOff); // out[:,0] == cur_t
        }

        // gather A fragments per ISA 16-bit A 16x32 layout:
        //   lane half h, VGPR v -> K = (v<4?0:16) + 8h + (v&3)*2, M = lcol
        //   VGPRs 0-3 and 4-7 are each 16 contiguous bytes of Amat row lcol.
        Frag16 a0, a1;
        {
            int base = lcol * 8 + half; // 16-byte units: byte lcol*128 + half*16
            a0.q[0] = A128[base + 0];   // K =  8h .. 8h+7
            a0.q[1] = A128[base + 2];   // K = 16+8h .. 16+8h+7
            a1.q[0] = A128[base + 4];   // K = 32+8h ..
            a1.q[1] = A128[base + 6];   // K = 48+8h ..
        }

        // matmul + state update per N-tile
        #pragma unroll
        for (int n = 0; n < 4; ++n) {
            v8f s = {};
            s = __builtin_amdgcn_wmma_f32_16x16x32_bf16(
                    false, a0.v, false, Bf[0][n].v, (short)0, s, false, false);
            s = __builtin_amdgcn_wmma_f32_16x16x32_bf16(
                    false, a1.v, false, Bf[1][n].v, (short)0, s, false, false);
            int coln = n * 16 + lcol;
            #pragma unroll
            for (int r = 0; r < 8; ++r) {
                size_t idx = rowOff[r] + tOff + 1 + coln;
                float xj   = __builtin_nontemporal_load(in + idx);
                float pred = c[n][r] + dif[r] * s[r];
                float val  = (flg[r] == -1.0f) ? pred : xj;
                c[n][r] = val;
                __builtin_nontemporal_store(val, out + idx);
            }
        }
        // no trailing barrier needed: single wave per block, DS ops are
        // program-ordered; the top-of-loop __syncthreads fences ct/fl reuse.
    }
}

extern "C" void kernel_launch(void* const* d_in, const int* in_sizes, int n_in,
                              void* d_out, int out_size, void* d_ws, size_t ws_size,
                              hipStream_t stream) {
    const float* in    = (const float*)d_in[0];
    const float* h0    = (const float*)d_in[1];
    const float* trace = (const float*)d_in[2];
    const float* mu    = (const float*)d_in[3];
    const float* dp    = (const float*)d_in[4];
    float* out = (float*)d_out;

    const int B = in_sizes[1] / (MDIM + 1);           // h0 is (B, 65)
    const int T = in_sizes[0] / (B * (MDIM + 1));     // inputs is (B, T, 65)

    unsigned short* odotT = (unsigned short*)d_ws;    // 64*64 bf16 = 8 KB

    build_odot_kernel<<<1, 64, 0, stream>>>(mu, dp, odotT);
    rnn_scan_kernel<<<B / 16, 32, 0, stream>>>(in, h0, trace, odotT, out, T);
}
